// GraphSAGE_65781719106245
// MI455X (gfx1250) — matrix operand order repaired and checked by reading, not compiled, not run on hardware
//
#include <hip/hip_runtime.h>
#include <hip/hip_bf16.h>

#define N_NODES 10000
#define N_EDGES 640000
#define D_IN    128
#define D_HID   128
#define D_OUT   64
#define SCAN_T  1024

typedef __attribute__((ext_vector_type(4)))  __bf16 v4bf;
typedef __attribute__((ext_vector_type(8)))  __bf16 v8bf;
typedef __attribute__((ext_vector_type(16))) __bf16 v16bf;
typedef __attribute__((ext_vector_type(8)))  float  v8f;

// ---------------- utility kernels ----------------

__global__ void fill_i32(int* p, int n, int v) {
    int t = blockIdx.x * blockDim.x + threadIdx.x;
    if (t < n) p[t] = v;
}

__global__ void degree_count(const int* __restrict__ dst, int* __restrict__ deg, int E) {
    int t = blockIdx.x * blockDim.x + threadIdx.x;
    if (t < E) atomicAdd(&deg[dst[t]], 1);
}

// Single-workgroup exclusive scan over N node degrees -> row_start[N+1], cursor copy.
__global__ void scan_kernel(const int* __restrict__ deg, int* __restrict__ rs,
                            int* __restrict__ cursor, int n) {
    __shared__ int part[SCAN_T];
    const int t = threadIdx.x;
    const int chunk = (n + SCAN_T - 1) / SCAN_T;
    const int lo = t * chunk;
    const int hi = (lo + chunk < n) ? lo + chunk : n;
    int s = 0;
    for (int i = lo; i < hi; ++i) s += deg[i];
    part[t] = s;
    __syncthreads();
    if (t == 0) {
        int acc = 0;
        for (int i = 0; i < SCAN_T; ++i) { int v = part[i]; part[i] = acc; acc += v; }
        rs[n] = acc;                      // total edge count
    }
    __syncthreads();
    int acc = part[t];
    for (int i = lo; i < hi; ++i) {
        rs[i] = acc;
        cursor[i] = acc;
        acc += deg[i];
    }
}

__global__ void csr_fill(const int* __restrict__ src, const int* __restrict__ dst,
                         int* __restrict__ cursor, int* __restrict__ csr_src, int E) {
    int t = blockIdx.x * blockDim.x + threadIdx.x;
    if (t < E) {
        int p = atomicAdd(&cursor[dst[t]], 1);
        csr_src[p] = src[t];
    }
}

__global__ void cvt_f32_bf16(const float* __restrict__ in, __hip_bfloat16* __restrict__ out, int n) {
    int t = blockIdx.x * blockDim.x + threadIdx.x;
    if (t < n) out[t] = __float2bfloat16(in[t]);
}

// Pack row-major f32 W[K][N] into bf16 WMMA B-matrix per-lane layout:
// lane of tile (kt,nt) holds 16 bf16: i<8 -> W[kt*32 + (lane>>4)*8 + i][nt*16 + (lane&15)], i>=8 -> W[kt*32 + 16 + (lane>>4)*8 + (i-8)][nt*16 + (lane&15)].
__global__ void pack_weight(const float* __restrict__ W, __hip_bfloat16* __restrict__ Wp, int K, int N) {
    int t = blockIdx.x * blockDim.x + threadIdx.x;
    if (t >= K * N) return;
    int idx  = t & 15;
    int lane = (t >> 4) & 31;
    int tile = t >> 9;
    int NT   = N >> 4;
    int nt   = tile % NT;
    int kt   = tile / NT;
    int k = kt * 32 + ((idx < 8) ? ((lane >> 4) * 8 + idx) : (16 + (lane >> 4) * 8 + (idx - 8)));
    int n = nt * 16 + (lane & 15);
    Wp[t] = __float2bfloat16(W[(size_t)k * N + n]);
}

// ---------------- aggregation: one wave per node, CSR mean, zero atomics ----------------

__global__ void aggregate_kernel(const __hip_bfloat16* __restrict__ h,
                                 const int* __restrict__ rs, const int* __restrict__ csr_src,
                                 __hip_bfloat16* __restrict__ hn) {
    const int wave = (blockIdx.x * blockDim.x + threadIdx.x) >> 5;
    const int lane = threadIdx.x & 31;
    if (wave >= N_NODES) return;
    const int beg = rs[wave];
    const int end = rs[wave + 1];
    const __bf16* hb = (const __bf16*)h + lane * 4;    // lane owns features 4l..4l+3
    float acc0 = 0.f, acc1 = 0.f, acc2 = 0.f, acc3 = 0.f;
    for (int j = beg; j < end; ++j) {
        int s = csr_src[j];                            // wave-uniform, L2/scalar-cached
        v4bf hv = *(const v4bf*)(hb + (size_t)s * D_IN);
        acc0 += (float)hv[0];
        acc1 += (float)hv[1];
        acc2 += (float)hv[2];
        acc3 += (float)hv[3];
    }
    const float inv = 1.0f / fmaxf((float)(end - beg), 1.0f);
    __hip_bfloat16* o = hn + (size_t)wave * D_IN + lane * 4;
    o[0] = __float2bfloat16(acc0 * inv);
    o[1] = __float2bfloat16(acc1 * inv);
    o[2] = __float2bfloat16(acc2 * inv);
    o[3] = __float2bfloat16(acc3 * inv);
}

// ---------------- WMMA GEMM: out = act(hs@Ws + hn@Wn + b) ----------------

__device__ __forceinline__ v16bf load_a16(const __bf16* p) {
    v8bf lo = *(const v8bf*)(p);
    v8bf hi = *(const v8bf*)(p + 16);
    v16bf r;
#pragma unroll
    for (int i = 0; i < 8; ++i) { r[i] = lo[i]; r[i + 8] = hi[i]; }
    return r;
}

template <bool LAST>
__global__ void sage_gemm(const __hip_bfloat16* __restrict__ hs_, const __hip_bfloat16* __restrict__ hn_,
                          const __hip_bfloat16* __restrict__ Wsp_, const __hip_bfloat16* __restrict__ Wnp_,
                          const float* __restrict__ bias, int d_in, int d_out,
                          __hip_bfloat16* __restrict__ out_bf, float* __restrict__ out_f) {
    const int lane  = threadIdx.x & 31;
    const int nt    = threadIdx.x >> 5;   // one wave per 16-wide output-feature tile
    const int mt    = blockIdx.x;         // one block per 16-node tile (waves share A rows)
    const int NT    = d_out >> 4;
    const int KT    = d_in >> 5;
    const int rlane = lane & 15;
    const int half  = lane >> 4;
    const int row   = mt * 16 + rlane;    // N_NODES % 16 == 0

    const __bf16* hs = (const __bf16*)hs_ + (size_t)row * d_in + half * 8;
    const __bf16* hn = (const __bf16*)hn_ + (size_t)row * d_in + half * 8;
    const __bf16* ws = (const __bf16*)Wsp_ + ((size_t)nt * 32 + lane) * 16;
    const __bf16* wn = (const __bf16*)Wnp_ + ((size_t)nt * 32 + lane) * 16;
    const size_t wstep = (size_t)NT * 512;

    v8f acc = {};
    for (int kt = 0; kt < KT; ++kt) {
        v16bf a0 = load_a16(hs + kt * 32);
        v16bf b0 = *(const v16bf*)(ws + kt * wstep);
        acc = __builtin_amdgcn_wmma_f32_16x16x32_bf16(false, a0, false, b0, (short)0, acc, false, false);
        v16bf a1 = load_a16(hn + kt * 32);
        v16bf b1 = *(const v16bf*)(wn + kt * wstep);
        acc = __builtin_amdgcn_wmma_f32_16x16x32_bf16(false, a1, false, b1, (short)0, acc, false, false);
    }

    const int ncol = nt * 16 + rlane;
    const float bv = bias[ncol];
#pragma unroll
    for (int i = 0; i < 8; ++i) {
        // C/D layout: VGPR i -> M = i + half*8, N = lane&15
        float v = acc[i] + bv;
        if (!LAST) v = fmaxf(v, 0.0f);
        int m = mt * 16 + half * 8 + i;
        if (LAST) out_f[(size_t)m * d_out + ncol] = v;
        else      out_bf[(size_t)m * d_out + ncol] = __float2bfloat16(v);
    }
}

// ---------------- host ----------------

extern "C" void kernel_launch(void* const* d_in, const int* in_sizes, int n_in,
                              void* d_out, int out_size, void* d_ws, size_t ws_size,
                              hipStream_t stream) {
    const float* inputs   = (const float*)d_in[0];
    const float* W_self0  = (const float*)d_in[1];
    const float* W_neigh0 = (const float*)d_in[2];
    const float* b0       = (const float*)d_in[3];
    const float* W_self1  = (const float*)d_in[4];
    const float* W_neigh1 = (const float*)d_in[5];
    const float* b1       = (const float*)d_in[6];
    const float* W_self2  = (const float*)d_in[7];
    const float* W_neigh2 = (const float*)d_in[8];
    const float* b2       = (const float*)d_in[9];
    const int*   esrc     = (const int*)d_in[10];
    const int*   edst     = (const int*)d_in[11];
    float* out = (float*)d_out;

    char* ws = (char*)d_ws;
    size_t off = 0;
    auto alloc = [&](size_t bytes) -> void* {
        void* p = (void*)(ws + off);
        off += (bytes + 255) & ~(size_t)255;
        return p;
    };
    int* deg     = (int*)alloc((size_t)N_NODES * sizeof(int));
    int* rs      = (int*)alloc((size_t)(N_NODES + 1) * sizeof(int));
    int* cursor  = (int*)alloc((size_t)N_NODES * sizeof(int));
    int* csr_src = (int*)alloc((size_t)N_EDGES * sizeof(int));
    __hip_bfloat16* hA = (__hip_bfloat16*)alloc((size_t)N_NODES * D_IN * 2);
    __hip_bfloat16* hB = (__hip_bfloat16*)alloc((size_t)N_NODES * D_IN * 2);
    __hip_bfloat16* hn = (__hip_bfloat16*)alloc((size_t)N_NODES * D_IN * 2);
    const float* Wsrc[6] = {W_self0, W_neigh0, W_self1, W_neigh1, W_self2, W_neigh2};
    const int    Wk[6]   = {D_IN, D_IN, D_HID, D_HID, D_HID, D_HID};
    const int    Wn[6]   = {D_HID, D_HID, D_HID, D_HID, D_OUT, D_OUT};
    __hip_bfloat16* Wp[6];
    for (int i = 0; i < 6; ++i)
        Wp[i] = (__hip_bfloat16*)alloc((size_t)Wk[i] * Wn[i] * 2);

    const int B = 256;
    const int nFeat = N_NODES * D_IN;

    // ---- CSR build (once per launch) ----
    fill_i32<<<(N_NODES + B - 1) / B, B, 0, stream>>>(deg, N_NODES, 0);
    degree_count<<<(N_EDGES + B - 1) / B, B, 0, stream>>>(edst, deg, N_EDGES);
    scan_kernel<<<1, SCAN_T, 0, stream>>>(deg, rs, cursor, N_NODES);
    csr_fill<<<(N_EDGES + B - 1) / B, B, 0, stream>>>(esrc, edst, cursor, csr_src, N_EDGES);

    // ---- input features -> bf16, weights -> packed bf16 WMMA-B layout ----
    cvt_f32_bf16<<<(nFeat + B - 1) / B, B, 0, stream>>>(inputs, hA, nFeat);
    for (int i = 0; i < 6; ++i) {
        int tot = Wk[i] * Wn[i];
        pack_weight<<<(tot + B - 1) / B, B, 0, stream>>>(Wsrc[i], Wp[i], Wk[i], Wn[i]);
    }

    const int aggBlocks = (N_NODES * 32 + B - 1) / B;   // one wave per node

    // layer 0: 128 -> 128, ReLU
    aggregate_kernel<<<aggBlocks, B, 0, stream>>>(hA, rs, csr_src, hn);
    sage_gemm<false><<<N_NODES / 16, 32 * (D_HID / 16), 0, stream>>>(
        hA, hn, Wp[0], Wp[1], b0, D_IN, D_HID, hB, nullptr);

    // layer 1: 128 -> 128, ReLU
    aggregate_kernel<<<aggBlocks, B, 0, stream>>>(hB, rs, csr_src, hn);
    sage_gemm<false><<<N_NODES / 16, 32 * (D_HID / 16), 0, stream>>>(
        hB, hn, Wp[2], Wp[3], b1, D_HID, D_HID, hA, nullptr);

    // layer 2: 128 -> 64, no activation, f32 output
    aggregate_kernel<<<aggBlocks, B, 0, stream>>>(hA, rs, csr_src, hn);
    sage_gemm<true><<<N_NODES / 16, 32 * (D_OUT / 16), 0, stream>>>(
        hA, hn, Wp[4], Wp[5], b2, D_HID, D_OUT, nullptr, out);
}
